// StandardAttention_13537736917789
// MI455X (gfx1250) — matrix operand order repaired and checked by reading, not compiled, not run on hardware
//
#include <hip/hip_runtime.h>
#include <math.h>

// ---------------- CDNA5 WMMA types ----------------
typedef __attribute__((ext_vector_type(16))) __bf16 v16bf;
typedef __attribute__((ext_vector_type(8)))  float  v8f;
typedef unsigned short u16;
typedef unsigned int   u32;

struct U4x2 { uint4 a, b; };

#define T_SEQ 4096
#define D_MODEL 1024
#define N_HEADS 16
#define D_HEAD 64

__device__ __forceinline__ u16 f2bf(float f) {
  unsigned u = __float_as_uint(f);
  u += 0x7FFFu + ((u >> 16) & 1u);   // round to nearest even
  return (u16)(u >> 16);
}

__device__ __forceinline__ u32 pack2bf(float lo, float hi) {
  return (u32)f2bf(lo) | ((u32)f2bf(hi) << 16);
}

__device__ __forceinline__ v8f vzero8() {
  v8f z;
#pragma unroll
  for (int i = 0; i < 8; ++i) z[i] = 0.0f;
  return z;
}

// A-matrix fragment (16x32 bf16, row-major source, ld in elements).
// Layout: lane(m,hi): e0..15 -> K = hi*16 + e (contiguous, matches load_b)
__device__ __forceinline__ v16bf load_a(const u16* base, int ld, int lane) {
  int m = lane & 15, hi = lane >> 4;
  const u16* p = base + m * ld + hi * 16;
  U4x2 t;
  t.a = *(const uint4*)(p);
  t.b = *(const uint4*)(p + 8);
  return __builtin_bit_cast(v16bf, t);
}

// B-matrix fragment (32x16 bf16). Column n contiguous along K with stride ldcol.
// Layout (contiguous, matches load_a): lane(n,hi): e0..15 -> K = hi*16 + e (32 contiguous bytes)
__device__ __forceinline__ v16bf load_b(const u16* base, int ldcol, int lane) {
  int n = lane & 15, hi = lane >> 4;
  const u16* p = base + n * ldcol + hi * 16;
  U4x2 t;
  t.a = *(const uint4*)(p);
  t.b = *(const uint4*)(p + 8);
  return __builtin_bit_cast(v16bf, t);
}

__device__ __forceinline__ v8f wmma_bf(v16bf a, v16bf b, v8f c) {
  return __builtin_amdgcn_wmma_f32_16x16x32_bf16(false, a, false, b, (short)0, c,
                                                 false, false);
}

// ---------------- fp32 -> bf16 conversion (vectorized x4) ----------------
__global__ void cvt_bf16x4(const float4* __restrict__ s, ushort4* __restrict__ d,
                           int n4) {
  int i = blockIdx.x * blockDim.x + threadIdx.x;
  int stride = gridDim.x * blockDim.x;
  for (; i < n4; i += stride) {
    float4 v = s[i];
    ushort4 o;
    o.x = f2bf(v.x); o.y = f2bf(v.y); o.z = f2bf(v.z); o.w = f2bf(v.w);
    d[i] = o;
  }
}

// ---------------- QKV projection ----------------
// out[t,e] = sum_d x[t,d]*qkv_w[e,d] + qkv_b[e];  e = s*1024 + h*64 + d
// 16(t) x 64(e) tile per wave; 2x-unrolled ping-pong K loop (no reg copies).
__global__ __launch_bounds__(32) void qkv_gemm(
    const u16* __restrict__ xb, const u16* __restrict__ wb,
    const float* __restrict__ bias, u16* __restrict__ Qb, u16* __restrict__ Kb,
    u16* __restrict__ VTb) {
  const int t0 = blockIdx.x * 16;
  const int e0 = blockIdx.y * 64;
  const int lane = threadIdx.x;
  const u16* abase = xb + (size_t)t0 * D_MODEL;
  const u16* wbase = wb + (size_t)e0 * D_MODEL;

  v8f acc[4];
#pragma unroll
  for (int j = 0; j < 4; ++j) acc[j] = vzero8();

  v16bf a0 = load_a(abase, D_MODEL, lane);
  v16bf b0_0 = load_b(wbase + 0 * 16 * D_MODEL, D_MODEL, lane);
  v16bf b0_1 = load_b(wbase + 1 * 16 * D_MODEL, D_MODEL, lane);
  v16bf b0_2 = load_b(wbase + 2 * 16 * D_MODEL, D_MODEL, lane);
  v16bf b0_3 = load_b(wbase + 3 * 16 * D_MODEL, D_MODEL, lane);

  for (int k = 0; k < D_MODEL; k += 64) {
    // load buffer-1 (k+32) while computing with buffer-0 (k)
    const int k1 = k + 32;
    v16bf a1 = load_a(abase + k1, D_MODEL, lane);
    v16bf b1_0 = load_b(wbase + 0 * 16 * D_MODEL + k1, D_MODEL, lane);
    v16bf b1_1 = load_b(wbase + 1 * 16 * D_MODEL + k1, D_MODEL, lane);
    v16bf b1_2 = load_b(wbase + 2 * 16 * D_MODEL + k1, D_MODEL, lane);
    v16bf b1_3 = load_b(wbase + 3 * 16 * D_MODEL + k1, D_MODEL, lane);
    __builtin_prefetch(abase + ((k + 128) & (D_MODEL - 1)), 0, 1);
    acc[0] = wmma_bf(a0, b0_0, acc[0]);
    acc[1] = wmma_bf(a0, b0_1, acc[1]);
    acc[2] = wmma_bf(a0, b0_2, acc[2]);
    acc[3] = wmma_bf(a0, b0_3, acc[3]);

    // load buffer-0 (k+64, wraps dead on last iter) while computing buffer-1
    const int k2 = (k + 64) & (D_MODEL - 1);
    a0 = load_a(abase + k2, D_MODEL, lane);
    b0_0 = load_b(wbase + 0 * 16 * D_MODEL + k2, D_MODEL, lane);
    b0_1 = load_b(wbase + 1 * 16 * D_MODEL + k2, D_MODEL, lane);
    b0_2 = load_b(wbase + 2 * 16 * D_MODEL + k2, D_MODEL, lane);
    b0_3 = load_b(wbase + 3 * 16 * D_MODEL + k2, D_MODEL, lane);
    acc[0] = wmma_bf(a1, b1_0, acc[0]);
    acc[1] = wmma_bf(a1, b1_1, acc[1]);
    acc[2] = wmma_bf(a1, b1_2, acc[2]);
    acc[3] = wmma_bf(a1, b1_3, acc[3]);
  }

  const int n = lane & 15, hi = lane >> 4;
  const int s = e0 >> 10;
  const int h = (e0 & 1023) >> 6;
#pragma unroll
  for (int j = 0; j < 4; ++j) {
#pragma unroll
    for (int r = 0; r < 8; ++r) {
      int d = j * 16 + n;
      int t = t0 + r + 8 * hi;
      float v = acc[j][r] + bias[e0 + d];
      u16 bv = f2bf(v);
      if (s == 0)      Qb[(h * T_SEQ + t) * D_HEAD + d] = bv;
      else if (s == 1) Kb[(h * T_SEQ + t) * D_HEAD + d] = bv;
      else             VTb[((size_t)h * D_HEAD + d) * T_SEQ + t] = bv;  // V^T
    }
  }
}

// ---------------- causal flash attention (transposed form) ----------------
// One wave: 16 queries of one head. S^T = K*Q^T puts each query in one lane
// (softmax = in-lane reduce + 1 cross-half shfl); O^T = V^T * P^T with the
// P^T B-fragment assembled in registers via one half-swap shuffle.
// K fragments are pipelined one block ahead so score WMMAs never wait.
__global__ __launch_bounds__(32) void attn_kernel(
    const u16* __restrict__ Qb, const u16* __restrict__ Kb,
    const u16* __restrict__ VTb, u16* __restrict__ AO) {
  const int q0 = blockIdx.x * 16;
  const int h = blockIdx.y;
  const int lane = threadIdx.x;
  const int n = lane & 15, hi = lane >> 4;

  const u16* Qh = Qb + (size_t)h * T_SEQ * D_HEAD;
  const u16* Kh = Kb + (size_t)h * T_SEQ * D_HEAD;
  const u16* Vh = VTb + (size_t)h * D_HEAD * T_SEQ;

  // Q^T as B fragments (cols = queries, K-dim = d), loaded once.
  v16bf bq0 = load_b(Qh + (size_t)q0 * D_HEAD + 0, D_HEAD, lane);
  v16bf bq1 = load_b(Qh + (size_t)q0 * D_HEAD + 32, D_HEAD, lane);

  v8f acc[4];   // O^T: lane n = query col, rows d = j*16 + r + 8*hi
#pragma unroll
  for (int j = 0; j < 4; ++j) acc[j] = vzero8();

  const float NEGI = -3.0e38f;
  float rm = NEGI, rs = 0.0f;               // per-query scalars (lane-local)
  const int q = q0 + n;                     // this lane's query index
  const float sc = 0.125f * 1.44269504088896f;  // 1/sqrt(64) * log2(e)
  const int kend = q0 + 16;

  // preload K fragments for the first key block
  v16bf kc0 = load_a(Kh + 0, D_HEAD, lane);
  v16bf kc1 = load_a(Kh + 32, D_HEAD, lane);
  v16bf kc2 = load_a(Kh + 16 * D_HEAD + 0, D_HEAD, lane);
  v16bf kc3 = load_a(Kh + 16 * D_HEAD + 32, D_HEAD, lane);

  for (int kb = 0; kb < kend; kb += 32) {
    // ---- issue loads: V for this block, K for the next block ----
    const int kbn = (kb + 32 < kend) ? kb + 32 : kb;  // dead reload on last
    v16bf av0 = load_a(Vh + (size_t)(0 * 16) * T_SEQ + kb, T_SEQ, lane);
    v16bf av1 = load_a(Vh + (size_t)(1 * 16) * T_SEQ + kb, T_SEQ, lane);
    v16bf av2 = load_a(Vh + (size_t)(2 * 16) * T_SEQ + kb, T_SEQ, lane);
    v16bf av3 = load_a(Vh + (size_t)(3 * 16) * T_SEQ + kb, T_SEQ, lane);
    v16bf kn0 = load_a(Kh + (size_t)kbn * D_HEAD + 0, D_HEAD, lane);
    v16bf kn1 = load_a(Kh + (size_t)kbn * D_HEAD + 32, D_HEAD, lane);
    v16bf kn2 = load_a(Kh + (size_t)(kbn + 16) * D_HEAD + 0, D_HEAD, lane);
    v16bf kn3 = load_a(Kh + (size_t)(kbn + 16) * D_HEAD + 32, D_HEAD, lane);

    // ---- S^T tiles (K frags already resident -> no wait) ----
    v8f s0 = vzero8(), s1 = vzero8();
    s0 = wmma_bf(kc0, bq0, s0);
    s0 = wmma_bf(kc1, bq1, s0);
    s1 = wmma_bf(kc2, bq0, s1);
    s1 = wmma_bf(kc3, bq1, s1);

    // ---- scale + causal mask (key > query -> -inf) ----
    const bool needmask = (kb + 31 > q0);
    float tmax = NEGI;
#pragma unroll
    for (int r = 0; r < 8; ++r) {
      s0[r] *= sc;
      s1[r] *= sc;
      if (needmask) {
        int key0 = kb + r + 8 * hi;
        if (key0 > q)      s0[r] = NEGI;
        if (key0 + 16 > q) s1[r] = NEGI;
      }
      tmax = fmaxf(tmax, fmaxf(s0[r], s1[r]));
    }
    tmax = fmaxf(tmax, __shfl_xor(tmax, 16, 32));  // combine key halves

    // ---- online softmax (scalars per lane/query) ----
    float nm = fmaxf(rm, tmax);
    float fac = exp2f(rm - nm);
    rm = nm;

    float p0[8], p1[8], ts = 0.0f;
#pragma unroll
    for (int r = 0; r < 8; ++r) {
      p0[r] = exp2f(s0[r] - nm);
      p1[r] = exp2f(s1[r] - nm);
      ts += p0[r] + p1[r];
    }
    ts += __shfl_xor(ts, 16, 32);
    rs = rs * fac + ts;
#pragma unroll
    for (int j = 0; j < 4; ++j)
#pragma unroll
      for (int r = 0; r < 8; ++r) acc[j][r] *= fac;

    // ---- assemble P^T B-fragment in registers ----
    // own u32 i = bf16 pair of keys (kb + 8*hi + 2i, +2i+1); half-swap partner
    u32 o0[4], o1[4], pr0[4], pr1[4];
#pragma unroll
    for (int i = 0; i < 4; ++i) {
      o0[i] = pack2bf(p0[2 * i], p0[2 * i + 1]);
      o1[i] = pack2bf(p1[2 * i], p1[2 * i + 1]);
    }
#pragma unroll
    for (int i = 0; i < 4; ++i) {
      pr0[i] = __shfl_xor(o0[i], 16, 32);
      pr1[i] = __shfl_xor(o1[i], 16, 32);
    }
    U4x2 t;
    if (hi == 0) {  // lane needs keys kb+0..15 : own(0..7) | partner(8..15)
      t.a = make_uint4(o0[0], o0[1], o0[2], o0[3]);
      t.b = make_uint4(pr0[0], pr0[1], pr0[2], pr0[3]);
    } else {        // lane needs keys kb+16..31 : partner(16..23) | own(24..31)
      t.a = make_uint4(pr1[0], pr1[1], pr1[2], pr1[3]);
      t.b = make_uint4(o1[0], o1[1], o1[2], o1[3]);
    }
    v16bf bP = __builtin_bit_cast(v16bf, t);

    // ---- O^T += V^T * P^T ----
    acc[0] = wmma_bf(av0, bP, acc[0]);
    acc[1] = wmma_bf(av1, bP, acc[1]);
    acc[2] = wmma_bf(av2, bP, acc[2]);
    acc[3] = wmma_bf(av3, bP, acc[3]);

    // rotate pipelined K fragments
    kc0 = kn0; kc1 = kn1; kc2 = kn2; kc3 = kn3;
  }

  // ---- normalize and store bf16 attention output [t][h*64+d] ----
  const float inv = 1.0f / rs;
#pragma unroll
  for (int j = 0; j < 4; ++j) {
#pragma unroll
    for (int r = 0; r < 8; ++r) {
      int d = j * 16 + r + 8 * hi;
      AO[(size_t)q * D_MODEL + h * D_HEAD + d] = f2bf(acc[j][r] * inv);
    }
  }
}

// ---------------- output projection ----------------
__global__ __launch_bounds__(32) void proj_gemm(
    const u16* __restrict__ AO, const u16* __restrict__ wb,
    const float* __restrict__ bias, float* __restrict__ out) {
  const int t0 = blockIdx.x * 16;
  const int e0 = blockIdx.y * 64;
  const int lane = threadIdx.x;
  const u16* abase = AO + (size_t)t0 * D_MODEL;
  const u16* wbase = wb + (size_t)e0 * D_MODEL;

  v8f acc[4];
#pragma unroll
  for (int j = 0; j < 4; ++j) acc[j] = vzero8();

  v16bf a0 = load_a(abase, D_MODEL, lane);
  v16bf b0_0 = load_b(wbase + 0 * 16 * D_MODEL, D_MODEL, lane);
  v16bf b0_1 = load_b(wbase + 1 * 16 * D_MODEL, D_MODEL, lane);
  v16bf b0_2 = load_b(wbase + 2 * 16 * D_MODEL, D_MODEL, lane);
  v16bf b0_3 = load_b(wbase + 3 * 16 * D_MODEL, D_MODEL, lane);

  for (int k = 0; k < D_MODEL; k += 64) {
    const int k1 = k + 32;
    v16bf a1 = load_a(abase + k1, D_MODEL, lane);
    v16bf b1_0 = load_b(wbase + 0 * 16 * D_MODEL + k1, D_MODEL, lane);
    v16bf b1_1 = load_b(wbase + 1 * 16 * D_MODEL + k1, D_MODEL, lane);
    v16bf b1_2 = load_b(wbase + 2 * 16 * D_MODEL + k1, D_MODEL, lane);
    v16bf b1_3 = load_b(wbase + 3 * 16 * D_MODEL + k1, D_MODEL, lane);
    __builtin_prefetch(abase + ((k + 128) & (D_MODEL - 1)), 0, 1);
    acc[0] = wmma_bf(a0, b0_0, acc[0]);
    acc[1] = wmma_bf(a0, b0_1, acc[1]);
    acc[2] = wmma_bf(a0, b0_2, acc[2]);
    acc[3] = wmma_bf(a0, b0_3, acc[3]);

    const int k2 = (k + 64) & (D_MODEL - 1);
    a0 = load_a(abase + k2, D_MODEL, lane);
    b0_0 = load_b(wbase + 0 * 16 * D_MODEL + k2, D_MODEL, lane);
    b0_1 = load_b(wbase + 1 * 16 * D_MODEL + k2, D_MODEL, lane);
    b0_2 = load_b(wbase + 2 * 16 * D_MODEL + k2, D_MODEL, lane);
    b0_3 = load_b(wbase + 3 * 16 * D_MODEL + k2, D_MODEL, lane);
    acc[0] = wmma_bf(a1, b1_0, acc[0]);
    acc[1] = wmma_bf(a1, b1_1, acc[1]);
    acc[2] = wmma_bf(a1, b1_2, acc[2]);
    acc[3] = wmma_bf(a1, b1_3, acc[3]);
  }

  const int n = lane & 15, hi = lane >> 4;
#pragma unroll
  for (int j = 0; j < 4; ++j) {
#pragma unroll
    for (int r = 0; r < 8; ++r) {
      int t = t0 + r + 8 * hi;
      int e = e0 + j * 16 + n;
      out[(size_t)t * D_MODEL + e] = acc[j][r] + bias[e];
    }
  }
}

// ---------------- host launcher ----------------
extern "C" void kernel_launch(void* const* d_in, const int* in_sizes, int n_in,
                              void* d_out, int out_size, void* d_ws, size_t ws_size,
                              hipStream_t stream) {
  (void)in_sizes; (void)n_in; (void)out_size; (void)ws_size;
  const float* x     = (const float*)d_in[0];
  const float* qkv_w = (const float*)d_in[1];
  const float* qkv_b = (const float*)d_in[2];
  const float* out_w = (const float*)d_in[3];
  const float* out_b = (const float*)d_in[4];
  float* out = (float*)d_out;

  char* ws = (char*)d_ws;
  u16* xb  = (u16*)(ws + 0);           //  8 MB  x in bf16
  u16* wqb = (u16*)(ws + 8388608);     //  6 MB  qkv_w bf16
  u16* wob = (u16*)(ws + 14680064);    //  2 MB  out_w bf16
  u16* Qb  = (u16*)(ws + 16777216);    //  8 MB  Q [h][t][d]
  u16* Kb  = (u16*)(ws + 25165824);    //  8 MB  K [h][t][d]
  u16* VTb = (u16*)(ws + 33554432);    //  8 MB  V^T [h][d][t]
  u16* AO  = (u16*)(ws + 41943040);    //  8 MB  attention out bf16

  cvt_bf16x4<<<1024, 256, 0, stream>>>((const float4*)x, (ushort4*)xb,
                                       T_SEQ * D_MODEL / 4);
  cvt_bf16x4<<<1024, 256, 0, stream>>>((const float4*)qkv_w, (ushort4*)wqb,
                                       3 * D_MODEL * D_MODEL / 4);
  cvt_bf16x4<<<512, 256, 0, stream>>>((const float4*)out_w, (ushort4*)wob,
                                      D_MODEL * D_MODEL / 4);

  qkv_gemm<<<dim3(T_SEQ / 16, 3 * D_MODEL / 64), 32, 0, stream>>>(
      xb, wqb, qkv_b, Qb, Kb, VTb);

  attn_kernel<<<dim3(T_SEQ / 16, N_HEADS), 32, 0, stream>>>(Qb, Kb, VTb, AO);

  proj_gemm<<<dim3(T_SEQ / 16, D_MODEL / 64), 32, 0, stream>>>(AO, wob, out_b, out);
}
